// GATv2Layer_30803505447250
// MI455X (gfx1250) — matrix operand order repaired and checked by reading, not compiled, not run on hardware
//
#include <hip/hip_runtime.h>

typedef __attribute__((ext_vector_type(16))) _Float16 v16h;
typedef __attribute__((ext_vector_type(8)))  _Float16 v8h;
typedef __attribute__((ext_vector_type(8)))  float    v8f;
typedef __attribute__((ext_vector_type(4)))  int      v4i;
typedef __attribute__((ext_vector_type(4)))  float    v4f;
typedef _Float16 half_t;

#define N_NODES 8192
#define F_INC   256
#define F_OUTC  128
#define NCHUNK  (N_NODES / 32)
#define BPITCH  40   // LDS row pitch in halfs (80B): bank-conflict-free, 16B aligned

// ---------------------------------------------------------------------------
// lane broadcast via ds_bpermute (wave32)
// ---------------------------------------------------------------------------
__device__ __forceinline__ float lane_bcast(float v, int srcLane) {
  int b = __builtin_amdgcn_ds_bpermute(srcLane << 2, __builtin_bit_cast(int, v));
  return __builtin_bit_cast(float, b);
}

// ---------------------------------------------------------------------------
// K1: h = x @ W  (f32 -> f16 WMMA, f32 accum), store h transposed as f16.
// One wave per 16x16 tile of h; 8 WMMAs over K=256.
// ---------------------------------------------------------------------------
__global__ __launch_bounds__(128)
void k1_feat(const float* __restrict__ x, const float* __restrict__ W,
             half_t* __restrict__ hT) {
  const int lane    = threadIdx.x & 31;
  const int wave    = threadIdx.x >> 5;
  const int tile    = blockIdx.x * 4 + wave;   // 0..4095
  const int rowblk  = tile >> 3;               // 0..511
  const int coltile = tile & 7;                // 0..7
  const int r16     = lane & 15;
  const int hi      = lane >> 4;

  const int row  = rowblk * 16 + r16;          // x row (A)
  const int ncol = coltile * 16 + r16;         // W col (B)

  v8f c = {};
  #pragma unroll
  for (int kc = 0; kc < F_INC / 32; ++kc) {
    const float* xp = x + (size_t)row * F_INC + kc * 32 + hi * 8;
    v4f l0 = *(const v4f*)(xp + 0);
    v4f l1 = *(const v4f*)(xp + 4);
    v4f h0 = *(const v4f*)(xp + 16);
    v4f h1 = *(const v4f*)(xp + 20);
    v16h a;
    a[0]=(half_t)l0[0]; a[1]=(half_t)l0[1]; a[2]=(half_t)l0[2]; a[3]=(half_t)l0[3];
    a[4]=(half_t)l1[0]; a[5]=(half_t)l1[1]; a[6]=(half_t)l1[2]; a[7]=(half_t)l1[3];
    a[8]=(half_t)h0[0]; a[9]=(half_t)h0[1]; a[10]=(half_t)h0[2]; a[11]=(half_t)h0[3];
    a[12]=(half_t)h1[0]; a[13]=(half_t)h1[1]; a[14]=(half_t)h1[2]; a[15]=(half_t)h1[3];
    const float* wp = W + (size_t)(kc * 32 + hi * 8) * F_OUTC + ncol;
    v16h b;
    #pragma unroll
    for (int i = 0; i < 16; ++i) b[i] = (half_t)wp[(size_t)(i + (i >= 8 ? 8 : 0)) * F_OUTC];
    c = __builtin_amdgcn_wmma_f32_16x16x32_f16(false, a, false, b,
                                               (short)0, c, false, false);
  }
  #pragma unroll
  for (int v = 0; v < 8; ++v) {
    int m = v + hi * 8;
    hT[(size_t)(coltile * 16 + r16) * N_NODES + rowblk * 16 + m] = (half_t)c[v];
  }
}

// ---------------------------------------------------------------------------
// K2: s1[r] = h[r]·a1, s2[r] = h[r]·a2 from hT (coalesced across lanes per col)
// ---------------------------------------------------------------------------
__global__ void k2_scores(const half_t* __restrict__ hT, const float* __restrict__ a,
                          float* __restrict__ s1, float* __restrict__ s2) {
  int r = blockIdx.x * blockDim.x + threadIdx.x;
  if (r >= N_NODES) return;
  float acc1 = 0.f, acc2 = 0.f;
  for (int c = 0; c < F_OUTC; ++c) {
    float hv = (float)hT[(size_t)c * N_NODES + r];
    acc1 += hv * a[c];
    acc2 += hv * a[F_OUTC + c];
  }
  s1[r] = acc1;
  s2[r] = acc2;
}

// ---------------------------------------------------------------------------
// K2b: max over s2 (single workgroup)
// ---------------------------------------------------------------------------
__global__ void k2b_max(const float* __restrict__ s2, float* __restrict__ maxs2) {
  __shared__ float red[256];
  float m = -3.4e38f;
  for (int i = threadIdx.x; i < N_NODES; i += 256) m = fmaxf(m, s2[i]);
  red[threadIdx.x] = m;
  __syncthreads();
  for (int s = 128; s > 0; s >>= 1) {
    if ((int)threadIdx.x < s) red[threadIdx.x] = fmaxf(red[threadIdx.x], red[threadIdx.x + s]);
    __syncthreads();
  }
  if (threadIdx.x == 0) *maxs2 = red[0];
}

// ---------------------------------------------------------------------------
// K3: fused masked-softmax attention + (P @ h) + ELU.
// - adj streamed exactly once with NT loads (256 MB > 192 MB L2: don't thrash).
// - hT chunk (32 x 128 f16 = 8 KB) staged in double-buffered LDS, shared by
//   the WG's 4 waves (4x dedup of B reads), software-pipelined.
// - Softmax stabilized with the upper bound M_i = leaky(s1_i + max_j s2_j)
//   (leaky_relu monotonic) -> single pass, no accumulator rescaling.
// ---------------------------------------------------------------------------
__global__ __launch_bounds__(128)
void k3_attn(const int* __restrict__ adj, const half_t* __restrict__ hT,
             const float* __restrict__ s1, const float* __restrict__ s2,
             const float* __restrict__ maxs2, float* __restrict__ out) {
  __shared__ half_t Bs[2][F_OUTC][BPITCH];     // 2 * 128 * 80B = 20 KB

  const int tid    = threadIdx.x;              // 0..127
  const int lane   = tid & 31;
  const int wave   = tid >> 5;
  const int rowblk = blockIdx.x * 4 + wave;    // 0..511
  const int r16    = lane & 15;
  const int hi     = lane >> 4;
  const int row    = rowblk * 16 + r16;        // this lane's logit row

  const float ms2 = *maxs2;
  const float s1r = s1[row];
  float eb = s1r + ms2;
  const float M = (eb > 0.f) ? eb : 0.2f * eb; // row-max upper bound

  v8f acc[8] = {};
  float lsum = 0.f;

  const size_t adjRow = (size_t)row * N_NODES;
  // each thread stages one column of h (64B per chunk) into LDS
  const v4f* gsrc = (const v4f*)(hT + (size_t)tid * N_NODES);

  // ---- prologue: stage chunk 0 into Bs[0]
  {
    v4f r0 = gsrc[0], r1 = gsrc[1], r2 = gsrc[2], r3 = gsrc[3];
    v4f* dst = (v4f*)&Bs[0][tid][0];
    dst[0] = r0; dst[1] = r1; dst[2] = r2; dst[3] = r3;
  }
  __syncthreads();

  for (int c = 0; c < NCHUNK; ++c) {
    const int p = c & 1;

    // ---- issue next chunk's staging loads early (overlap with compute)
    v4f r0, r1, r2, r3;
    if (c + 1 < NCHUNK) {
      const v4f* g = gsrc + (size_t)(c + 1) * 4;   // 64B per chunk
      r0 = g[0]; r1 = g[1]; r2 = g[2]; r3 = g[3];
    }

    const int kLow = c * 32 + hi * 8;

    // prefetch next chunk of this adj row (gfx1250 global_prefetch_b8)
    if (c + 1 < NCHUNK)
      __builtin_prefetch(adj + adjRow + kLow + 32, 0, 1);

    // adjacency (non-temporal: single-use 256 MB stream) + s2
    const v4i* ap = (const v4i*)(adj + adjRow + kLow);
    const v4f* fp = (const v4f*)(s2 + kLow);
    v4i q0 = __builtin_nontemporal_load(ap + 0);
    v4i q1 = __builtin_nontemporal_load(ap + 1);
    v4i q2 = __builtin_nontemporal_load(ap + 4);
    v4i q3 = __builtin_nontemporal_load(ap + 5);
    v4f f0 = fp[0], f1 = fp[1], f2 = fp[4], f3 = fp[5];

    int   mv[16] = { q0[0],q0[1],q0[2],q0[3], q1[0],q1[1],q1[2],q1[3],
                     q2[0],q2[1],q2[2],q2[3], q3[0],q3[1],q3[2],q3[3] };
    float sv[16] = { f0[0],f0[1],f0[2],f0[3], f1[0],f1[1],f1[2],f1[3],
                     f2[0],f2[1],f2[2],f2[3], f3[0],f3[1],f3[2],f3[3] };

    // p_ij = adj ? exp(leaky(s1+s2) - M) : 0   (masked entries exactly 0)
    v16h a;
    #pragma unroll
    for (int i = 0; i < 16; ++i) {
      float e = s1r + sv[i];
      e = (e > 0.f) ? e : 0.2f * e;
      float pv = (mv[i] > 0) ? __expf(e - M) : 0.f;
      lsum += pv;
      a[i] = (half_t)pv;
    }

    // acc[t] += P(16x32) @ h_chunk(32x16) from LDS, 8 output col-tiles
    #pragma unroll
    for (int t = 0; t < 8; ++t) {
      const half_t* bp = &Bs[p][t * 16 + r16][hi * 8];
      v8h b0 = *(const v8h*)(bp + 0);
      v8h b1 = *(const v8h*)(bp + 16);
      v16h b = __builtin_shufflevector(b0, b1, 0,1,2,3,4,5,6,7,
                                       8,9,10,11,12,13,14,15);
      acc[t] = __builtin_amdgcn_wmma_f32_16x16x32_f16(false, a, false, b,
                                                      (short)0, acc[t], false, false);
    }

    // ---- stage next chunk into the other buffer
    if (c + 1 < NCHUNK) {
      v4f* dst = (v4f*)&Bs[1 - p][tid][0];
      dst[0] = r0; dst[1] = r1; dst[2] = r2; dst[3] = r3;
    }
    __syncthreads();
  }

  // combine partial row sums: lanes L and L+16 hold the same logit row
  float ltot = lsum + lane_bcast(lsum, lane ^ 16);
  // lanes 0..15 (and 16..31) now hold l for rows 0..15

  #pragma unroll
  for (int v = 0; v < 8; ++v) {
    int m = v + hi * 8;                 // C-layout row of this VGPR
    float lm = lane_bcast(ltot, m);     // l for row m (from lane m)
    float inv = (lm != 0.f) ? (1.f / lm) : 0.f;
    #pragma unroll
    for (int t = 0; t < 8; ++t) {
      float hp = acc[t][v] * inv;
      float o = (hp > 0.f) ? hp : (__expf(hp) - 1.f);   // ELU(alpha=1)
      __builtin_nontemporal_store(
          o, out + (size_t)(rowblk * 16 + m) * F_OUTC + t * 16 + r16);
    }
  }
}

// ---------------------------------------------------------------------------
extern "C" void kernel_launch(void* const* d_in, const int* in_sizes, int n_in,
                              void* d_out, int out_size, void* d_ws, size_t ws_size,
                              hipStream_t stream) {
  const float* x   = (const float*)d_in[0];   // 8192 x 256
  const float* W   = (const float*)d_in[1];   // 256 x 128
  const float* a   = (const float*)d_in[2];   // 256 x 1
  const int*   adj = (const int*)d_in[3];     // 8192 x 8192
  float* out = (float*)d_out;                 // 8192 x 128

  char* ws = (char*)d_ws;
  half_t* hT  = (half_t*)ws;                                   // 2 MB
  float*  s1  = (float*)(ws + (size_t)F_OUTC * N_NODES * 2);   // 32 KB
  float*  s2  = s1 + N_NODES;                                  // 32 KB
  float*  ms2 = s2 + N_NODES;                                  // 4 B

  k1_feat  <<<(N_NODES / 16) * (F_OUTC / 16) / 4, 128, 0, stream>>>(x, W, hT);
  k2_scores<<<N_NODES / 256, 256, 0, stream>>>(hT, a, s1, s2);
  k2b_max  <<<1, 256, 0, stream>>>(s2, ms2);
  k3_attn  <<<(N_NODES / 16) / 4, 128, 0, stream>>>(adj, hT, s1, s2, ms2, out);
}
